// SparseDynamicVoxelAttention_38517266711010
// MI455X (gfx1250) — matrix operand order repaired and checked by reading, not compiled, not run on hardware
//
#include <hip/hip_runtime.h>
#include <hip/hip_bf16.h>

typedef __attribute__((ext_vector_type(16))) _Float16 v16h;
typedef __attribute__((ext_vector_type(8)))  _Float16 v8h;
typedef __attribute__((ext_vector_type(4)))  _Float16 v4h;
typedef __attribute__((ext_vector_type(8)))  float    v8f;
typedef __attribute__((ext_vector_type(4)))  float    v4f;
typedef __attribute__((ext_vector_type(4)))  unsigned v4u;
typedef __attribute__((ext_vector_type(8)))  unsigned v8u;

#define VOX   4096
#define BATCH 4
#define DIM   256
#define HID   64
#define KNN_K 10
#define TOPK  4
#define NHEAD 4
#define DHEAD 64
#define KK    256        // contraction dim for every GEMM in this model
#define WN    32         // per-wave N tile (2 x 16 WMMA subtiles)

// ---------------------------------------------------------------------------
// tok (f32) -> f16 once; A-side of all five GEMMs then reads pure f16.
// ---------------------------------------------------------------------------
__global__ void __launch_bounds__(256)
cvt_f32_to_f16(const float* __restrict__ in, _Float16* __restrict__ out)
{
  const long i = ((long)blockIdx.x * 256 + threadIdx.x) * 4;
  const v4f x = *(const v4f*)(in + i);
  v4h y;
  #pragma unroll
  for (int t = 0; t < 4; ++t) y[t] = (_Float16)x[t];
  *(v4h*)(out + i) = y;
}

// ---------------------------------------------------------------------------
// WMMA GEMM: C[b] = A[b] (M x 256 f16, row-major) @ W (256 x N f32 slice,
// row stride ldw) + bias.  256 threads = 8 waves; wave owns a 16x32 tile
// (2 WMMA subtiles sharing one A fragment); block tile = 128 x 32.
//
// Weight tile staged f32 -> LDS by the Tensor Data Mover (tensor_load_to_lds,
// D# per CDNA5 ISA ch.8), then converted once to f16 transposed [n][k] so B
// fragments are contiguous ds_load_b128 pairs.  K loop fully unrolled:
// 16 static v_wmma_f32_16x16x32_f16 per kernel.
// ---------------------------------------------------------------------------
__global__ void __launch_bounds__(256)
gemm_f16wmma(const _Float16* __restrict__ A, const float* __restrict__ W,
             const float* __restrict__ bias, float* __restrict__ C,
             int N, int ldw, long strideA, long strideC)
{
  const int wave  = threadIdx.x >> 5;
  const int lane  = threadIdx.x & 31;
  const int hl    = lane & 15;
  const int hi    = lane >> 4;
  const int tileM = (blockIdx.x * 8 + wave) * 16;
  const int tileN = blockIdx.y * WN;

  __shared__ float    sWf[KK * WN];      // 32 KB, TDM destination (row-major)
  __shared__ _Float16 sWh[WN * KK];      // 16 KB, transposed [n][k] f16

  // ---- TDM: DMA W[0:256, tileN:tileN+32] (f32, row stride ldw) into LDS ----
  if (__builtin_amdgcn_readfirstlane((int)threadIdx.x) == 0) {   // wave 0 only
    const unsigned long long ga = (unsigned long long)(const void*)(W + tileN);
    const unsigned ldsOff = (unsigned)(unsigned long long)(void*)&sWf[0];
    v4u g0; v8u g1; v4u g2 = {0, 0, 0, 0}; v4u g3 = {0, 0, 0, 0};
    g0[0] = 1u;                                  // count=1 (valid descriptor)
    g0[1] = ldsOff;                              // lds_addr
    g0[2] = (unsigned)ga;                        // global_addr[31:0]
    g0[3] = (unsigned)(ga >> 32) | (2u << 30);   // global_addr[56:32] | type=2
    g1[0] = 2u << 16;                            // data_size = 4 bytes
    g1[1] = (unsigned)WN << 16;                  // tensor_dim0 = 32
    g1[2] = (unsigned)KK << 16;                  // tensor_dim1 = 256
    g1[3] = (unsigned)WN << 16;                  // tile_dim0   = 32
    g1[4] = (unsigned)KK;                        // tile_dim1   = 256 (dim2=0)
    g1[5] = (unsigned)ldw;                       // tensor_dim0_stride[31:0]
    g1[6] = 0u;                                  // stride hi / dim1_stride lo
    g1[7] = 0u;
    asm volatile("tensor_load_to_lds %0, %1, %2, %3"
                 :: "s"(g0), "s"(g1), "s"(g2), "s"(g3) : "memory");
  }
  __builtin_amdgcn_s_wait_tensorcnt(0);
  __syncthreads();

  // ---- one-shot f32 -> f16 transpose into [n][k] layout ----
  for (int i = threadIdx.x; i < KK * WN; i += 256) {
    const int k = i >> 5, n = i & (WN - 1);      // sWf is [k][n]
    sWh[n * KK + k] = (_Float16)sWf[i];
  }
  __syncthreads();

  const float bl = bias ? bias[tileN + hl] : 0.0f;
  const float bh = bias ? bias[tileN + 16 + hl] : 0.0f;
  v8f acc0 = {bl, bl, bl, bl, bl, bl, bl, bl};
  v8f acc1 = {bh, bh, bh, bh, bh, bh, bh, bh};

  const int kbA = hi * 8;     // A: K split {0..7,16..23} / {8..15,24..31}
  const int kbB = hi * 8;     // B: K split {0..7,16..23} / {8..15,24..31}
  const _Float16* arow  = A + (long)blockIdx.z * strideA + (long)(tileM + hl) * KK;
  const _Float16* brow0 = sWh + hl * KK;
  const _Float16* brow1 = sWh + (hl + 16) * KK;

  #pragma unroll
  for (int k0 = 0; k0 < KK; k0 += 32) {
    __builtin_prefetch(arow + k0 + 128, 0, 1);   // global_prefetch_b8

    const v8h a0 = *(const v8h*)(arow + k0 + kbA);
    const v8h a1 = *(const v8h*)(arow + k0 + 16 + kbA);
    const v16h af = __builtin_shufflevector(a0, a1,
        0, 1, 2, 3, 4, 5, 6, 7, 8, 9, 10, 11, 12, 13, 14, 15);

    const v8h x0 = *(const v8h*)(brow0 + k0 + kbB);
    const v8h x1 = *(const v8h*)(brow0 + k0 + 16 + kbB);
    const v16h bf0 = __builtin_shufflevector(x0, x1,
        0, 1, 2, 3, 4, 5, 6, 7, 8, 9, 10, 11, 12, 13, 14, 15);

    const v8h y0 = *(const v8h*)(brow1 + k0 + kbB);
    const v8h y1 = *(const v8h*)(brow1 + k0 + 16 + kbB);
    const v16h bf1 = __builtin_shufflevector(y0, y1,
        0, 1, 2, 3, 4, 5, 6, 7, 8, 9, 10, 11, 12, 13, 14, 15);

    acc0 = __builtin_amdgcn_wmma_f32_16x16x32_f16(false, af, false, bf0,
                                                  (short)0, acc0, false, false);
    acc1 = __builtin_amdgcn_wmma_f32_16x16x32_f16(false, af, false, bf1,
                                                  (short)0, acc1, false, false);
  }

  // C/D layout: n = lane&15, m = r + 8*(lane>>4)
  float* crow = C + (long)blockIdx.z * strideC
                  + (long)(tileM + hi * 8) * N + tileN + hl;
  #pragma unroll
  for (int r = 0; r < 8; ++r) {
    crow[(long)r * N]      = acc0[r];
    crow[(long)r * N + 16] = acc1[r];
  }
}

// ---------------------------------------------------------------------------
// KNN: one thread per voxel, LDS-staged candidate tiles, fully-unrolled
// 11-entry sorted-insert register ladder.  Entry 0 is self; emit 1..10.
// ---------------------------------------------------------------------------
__global__ void __launch_bounds__(256)
knn_kernel(const float* __restrict__ coords, int* __restrict__ knn)
{
  const int b = blockIdx.y;
  const int v = blockIdx.x * 256 + threadIdx.x;
  const float* cb = coords + (long)b * VOX * 3;
  const float cx = cb[v * 3 + 0], cy = cb[v * 3 + 1], cz = cb[v * 3 + 2];

  float bd[KNN_K + 1];
  int   bi[KNN_K + 1];
  #pragma unroll
  for (int i = 0; i <= KNN_K; ++i) { bd[i] = 3.0e38f; bi[i] = 0; }

  __shared__ float scoo[256 * 3];
  for (int tile = 0; tile < VOX; tile += 256) {
    __syncthreads();
    const int j = tile + threadIdx.x;
    scoo[threadIdx.x * 3 + 0] = cb[j * 3 + 0];
    scoo[threadIdx.x * 3 + 1] = cb[j * 3 + 1];
    scoo[threadIdx.x * 3 + 2] = cb[j * 3 + 2];
    __syncthreads();
    for (int jj = 0; jj < 256; ++jj) {
      const float dx = scoo[jj * 3 + 0] - cx;
      const float dy = scoo[jj * 3 + 1] - cy;
      const float dz = scoo[jj * 3 + 2] - cz;
      float d = dx * dx + dy * dy + dz * dz;
      int idx = tile + jj;
      if (d < bd[KNN_K]) {
        #pragma unroll
        for (int i = 0; i <= KNN_K; ++i) {
          const bool c = d < bd[i];
          const float td = c ? bd[i] : d;
          const int   ti = c ? bi[i] : idx;
          bd[i] = c ? d : bd[i];
          bi[i] = c ? idx : bi[i];
          d = td; idx = ti;
        }
      }
    }
  }
  int* out = knn + ((long)b * VOX + v) * KNN_K;
  #pragma unroll
  for (int i = 1; i <= KNN_K; ++i) out[i - 1] = bi[i];
}

// ---------------------------------------------------------------------------
// Edge scoring + top-4 selection.  One wave32 per voxel; lane owns hidden
// dims {lane, lane+32}.  Softmax is monotone -> top-4 of raw scores.
// ---------------------------------------------------------------------------
__global__ void __launch_bounds__(256)
score_select(const float* __restrict__ Pa, const float* __restrict__ Pb,
             const float* __restrict__ coords,
             const float* __restrict__ W1c, const float* __restrict__ b1,
             const float* __restrict__ W2, const float* __restrict__ b2,
             const int* __restrict__ knn, int* __restrict__ sel)
{
  const int wave = threadIdx.x >> 5;
  const int lane = threadIdx.x & 31;
  const int b = blockIdx.y;
  const int v = blockIdx.x * 8 + wave;
  const long vb = (long)b * VOX + v;
  const float* cb = coords + (long)b * VOX * 3;

  const float a0 = Pa[vb * HID + lane];
  const float a1 = Pa[vb * HID + lane + 32];
  const float b1a = b1[lane],             b1b = b1[lane + 32];
  const float w2a = W2[lane],             w2b = W2[lane + 32];
  const float wcx0 = W1c[0 * HID + lane], wcx1 = W1c[0 * HID + lane + 32];
  const float wcy0 = W1c[1 * HID + lane], wcy1 = W1c[1 * HID + lane + 32];
  const float wcz0 = W1c[2 * HID + lane], wcz1 = W1c[2 * HID + lane + 32];
  const float cx = cb[v * 3], cy = cb[v * 3 + 1], cz = cb[v * 3 + 2];
  const float b2v = b2[0];

  int   nbIdx[KNN_K];
  float score[KNN_K];
  const int* kn = knn + vb * KNN_K;
  #pragma unroll
  for (int k = 0; k < KNN_K; ++k) {
    const int nb = kn[k];
    nbIdx[k] = nb;
    const float* pb = Pb + ((long)b * VOX + nb) * HID;
    const float rx = cb[nb * 3 + 0] - cx;
    const float ry = cb[nb * 3 + 1] - cy;
    const float rz = cb[nb * 3 + 2] - cz;
    float t0 = a0 + pb[lane]      + rx * wcx0 + ry * wcy0 + rz * wcz0 + b1a;
    float t1 = a1 + pb[lane + 32] + rx * wcx1 + ry * wcy1 + rz * wcz1 + b1b;
    t0 = fmaxf(t0, 0.0f); t1 = fmaxf(t1, 0.0f);
    float p = t0 * w2a + t1 * w2b;
    #pragma unroll
    for (int off = 16; off > 0; off >>= 1) p += __shfl_xor(p, off, 32);
    score[k] = p + b2v;
  }

  if (lane == 0) {
    unsigned used = 0;
    int* so = sel + vb * TOPK;
    #pragma unroll
    for (int t = 0; t < TOPK; ++t) {
      float best = -3.0e38f; int bj = 0;
      #pragma unroll
      for (int j = 0; j < KNN_K; ++j) {
        const bool better = (score[j] > best) && !((used >> j) & 1u);
        best = better ? score[j] : best;
        bj   = better ? j : bj;
      }
      used |= 1u << bj;
      so[t] = nbIdx[bj];
    }
  }
}

// ---------------------------------------------------------------------------
// Local attention over TOPK=4 selected neighbors.  One wave32 per
// (voxel, head); lane owns head dims {lane, lane+32}.  Emits f16 so the
// final @Wo GEMM consumes WMMA-ready A data directly.
// ---------------------------------------------------------------------------
__global__ void __launch_bounds__(256)
attention_kernel(const float* __restrict__ Q, const float* __restrict__ Kp,
                 const float* __restrict__ Vp, const int* __restrict__ sel,
                 _Float16* __restrict__ attout)
{
  const int lane = threadIdx.x & 31;
  const int wid = blockIdx.x * 8 + (threadIdx.x >> 5);
  const int v = wid >> 2;                    // NHEAD = 4
  const int h = wid & 3;
  const int b = blockIdx.y;
  const long vb = (long)b * VOX + v;

  const float* q = Q + vb * DIM + h * DHEAD;
  const float q0 = q[lane], q1 = q[lane + 32];

  int   nb[TOPK];
  float lg[TOPK];
  const int* sp = sel + vb * TOPK;
  #pragma unroll
  for (int s = 0; s < TOPK; ++s) {
    nb[s] = sp[s];
    const float* kk = Kp + ((long)b * VOX + nb[s]) * DIM + h * DHEAD;
    float p = q0 * kk[lane] + q1 * kk[lane + 32];
    #pragma unroll
    for (int off = 16; off > 0; off >>= 1) p += __shfl_xor(p, off, 32);
    lg[s] = p * 0.125f;                      // 1/sqrt(64)
  }
  const float mx = fmaxf(fmaxf(lg[0], lg[1]), fmaxf(lg[2], lg[3]));
  float e[TOPK], sum = 0.0f;
  #pragma unroll
  for (int s = 0; s < TOPK; ++s) { e[s] = __expf(lg[s] - mx); sum += e[s]; }
  const float inv = 1.0f / sum;

  float o0 = 0.0f, o1 = 0.0f;
  #pragma unroll
  for (int s = 0; s < TOPK; ++s) {
    const float w = e[s] * inv;
    const float* vv = Vp + ((long)b * VOX + nb[s]) * DIM + h * DHEAD;
    o0 += w * vv[lane];
    o1 += w * vv[lane + 32];
  }
  _Float16* op = attout + vb * DIM + h * DHEAD;
  op[lane]      = (_Float16)o0;
  op[lane + 32] = (_Float16)o1;
}

// ---------------------------------------------------------------------------
extern "C" void kernel_launch(void* const* d_in, const int* in_sizes, int n_in,
                              void* d_out, int out_size, void* d_ws, size_t ws_size,
                              hipStream_t stream)
{
  (void)in_sizes; (void)n_in; (void)out_size; (void)ws_size;

  const float* tok    = (const float*)d_in[0];
  const float* coords = (const float*)d_in[1];
  // d_in[2] = mask (all true, unused by reference math)
  const float* W1 = (const float*)d_in[3];
  const float* b1 = (const float*)d_in[4];
  const float* W2 = (const float*)d_in[5];
  const float* b2 = (const float*)d_in[6];
  const float* Wq = (const float*)d_in[7];
  const float* bq = (const float*)d_in[8];
  const float* Wk = (const float*)d_in[9];
  const float* bk = (const float*)d_in[10];
  const float* Wv = (const float*)d_in[11];
  const float* bv = (const float*)d_in[12];
  const float* Wo = (const float*)d_in[13];
  const float* bo = (const float*)d_in[14];
  float* out = (float*)d_out;

  char* ws = (char*)d_ws;
  size_t off = 0;
  auto alloc = [&](size_t bytes) -> void* {
    void* p = ws + off;
    off = (off + bytes + 255) & ~(size_t)255;
    return p;
  };
  int*      knn     = (int*)     alloc((size_t)BATCH * VOX * KNN_K * 4);
  int*      sel     = (int*)     alloc((size_t)BATCH * VOX * TOPK  * 4);
  _Float16* tokh    = (_Float16*)alloc((size_t)BATCH * VOX * DIM * 2);
  _Float16* attouth = (_Float16*)alloc((size_t)BATCH * VOX * DIM * 2);
  float*    Pa      = (float*)   alloc((size_t)BATCH * VOX * HID * 4);
  float*    Pb      = (float*)   alloc((size_t)BATCH * VOX * HID * 4);
  float*    Qb      = (float*)   alloc((size_t)BATCH * VOX * DIM * 4);
  float*    Kb      = (float*)   alloc((size_t)BATCH * VOX * DIM * 4);
  float*    Vb      = (float*)   alloc((size_t)BATCH * VOX * DIM * 4);

  const dim3 blk(256);
  const long sTok = (long)VOX * DIM;   // per-batch stride (elements)

  // 0) One-time f32 -> f16 conversion of tokens (A side of 5 GEMMs)
  cvt_f32_to_f16<<<dim3(BATCH * VOX * DIM / (256 * 4)), blk, 0, stream>>>(tok, tokh);

  // 1) KNN (independent)
  knn_kernel<<<dim3(VOX / 256, BATCH), blk, 0, stream>>>(coords, knn);

  // 2) Edge-MLP per-token projections: Pa = tok@W1[0:256], Pb = tok@W1[256:512]
  gemm_f16wmma<<<dim3(VOX / 128, HID / WN, BATCH), blk, 0, stream>>>(
      tokh, W1,             nullptr, Pa, HID, HID, sTok, (long)VOX * HID);
  gemm_f16wmma<<<dim3(VOX / 128, HID / WN, BATCH), blk, 0, stream>>>(
      tokh, W1 + 256 * HID, nullptr, Pb, HID, HID, sTok, (long)VOX * HID);

  // 3) Dense Q/K/V projections over ALL tokens (gathered later by sel)
  gemm_f16wmma<<<dim3(VOX / 128, DIM / WN, BATCH), blk, 0, stream>>>(
      tokh, Wq, bq, Qb, DIM, DIM, sTok, sTok);
  gemm_f16wmma<<<dim3(VOX / 128, DIM / WN, BATCH), blk, 0, stream>>>(
      tokh, Wk, bk, Kb, DIM, DIM, sTok, sTok);
  gemm_f16wmma<<<dim3(VOX / 128, DIM / WN, BATCH), blk, 0, stream>>>(
      tokh, Wv, bv, Vb, DIM, DIM, sTok, sTok);

  // 4) Edge scores + top-4 neighbor selection
  score_select<<<dim3(VOX / 8, BATCH), blk, 0, stream>>>(
      Pa, Pb, coords, W1 + 512 * HID, b1, W2, b2, knn, sel);

  // 5) 4-head attention over the 4 selected neighbors (f16 out)
  attention_kernel<<<dim3(VOX * NHEAD / 8, BATCH), blk, 0, stream>>>(
      Qb, Kb, Vb, sel, attouth);

  // 6) Output projection straight into d_out
  gemm_f16wmma<<<dim3(VOX / 128, DIM / WN, BATCH), blk, 0, stream>>>(
      attouth, Wo, bo, out, DIM, DIM, sTok, sTok);
}
